// Ansatz_45140106281626
// MI455X (gfx1250) — compile-verified, matrix operand order checked by reading, not run on hardware
//
#include <hip/hip_runtime.h>
#include <hip/hip_bf16.h>
#include <hip/hip_fp16.h>

typedef _Float16 half_t;
typedef __attribute__((ext_vector_type(16))) _Float16 v16h;
typedef __attribute__((ext_vector_type(8)))  _Float16 v8h;
typedef __attribute__((ext_vector_type(8)))  float    v8f;

namespace {
constexpr int NE    = 16;
constexpr int NU    = 8;
constexpr int NA    = 4;
constexpr int NSV   = 256;
constexpr int NPV   = 32;
constexpr int NFB   = 4;
constexpr int NDET  = 16;
constexpr int SVOUT = 3 * NSV + 2 * NPV;    // 832
constexpr int KT_SV = SVOUT / 32;           // 26
constexpr int WPB   = 2;                    // walkers per workgroup
constexpr int ROWS  = WPB * NE;             // 32
constexpr int PROWS = WPB * NE * NE;        // 512
constexpr int THREADS = 256;

// ---- LDS layout: blk first (32B aligned), then f32 arrays ----
constexpr int BLK_H  = 2 /*rowtiles*/ * KT_SV * 512;   // 26624 halfs
constexpr int SV_F   = ROWS * NSV;              // 8192
constexpr int P_F    = PROWS * NPV;             // 16384 (x2)
constexpr int PRAW_F = PROWS * 4;               // 2048
constexpr int ORB_F  = 2 * WPB * NDET * 64;     // 4096
constexpr int SMEM_FLOATS = SV_F + 2 * P_F + PRAW_F + ORB_F
                          + 2 * WPB * NSV /*mu+md*/ + 2 * WPB * NE * NPV /*pu+pd*/
                          + ROWS /*env*/ + ROWS * 3 /*rL*/ + NA * 3 /*aL*/
                          + 64 /*dsign*/ + 64 /*dld*/;
constexpr size_t SMEM_BYTES = size_t(BLK_H) * 2 + size_t(SMEM_FLOATS) * 4;

// ---- swizzled-weight workspace offsets (in halfs) ----
constexpr size_t WS_SW0 = 0;
constexpr size_t WS_SW  = WS_SW0 + 2 * 16 * 512;
constexpr size_t WS_VW  = WS_SW + size_t(4) * KT_SV * 16 * 512;
constexpr size_t WS_PW  = WS_VW + size_t(KT_SV) * 16 * 512;
constexpr size_t WS_WU  = WS_PW + 4 * 2 * 512;
constexpr size_t WS_WD  = WS_WU + 8 * 8 * 512;
constexpr size_t SW_LAYER = size_t(KT_SV) * 16 * 512;
} // namespace

// Force a 32-byte fragment load through the global address space (global_load_b128 x2).
__device__ __forceinline__ v16h ldg_v16h(const half_t* p) {
    typedef const __attribute__((address_space(1))) v16h* gp_t;
    return *(gp_t)(unsigned long long)p;
}

// Branchless tanh: exact saturation at +-1, v_exp_f32 based.
__device__ __forceinline__ float fast_tanh(float x) {
    float e = __expf(2.f * x);
    return 1.f - 2.f / (e + 1.f);
}

// K offset inside a 32-wide K tile for half-slot h of an f16 WMMA fragment.
__device__ __forceinline__ int fkoff(int lane, int h) {
    return ((h & 8) ? (h + 8) : h) + ((lane & 16) >> 1);
}

// Inverse mapping: (row, c) -> linear index in fragment-ordered blk.
// Layout: [rowtile][kt][lane][h], 512 halfs per fragment.
__device__ __forceinline__ int blk_index(int row, int c, int ktiles) {
    int rowtile = row >> 4, l15 = row & 15;
    int kt = c >> 5, koff = c & 31;
    int g = (koff >> 3) & 1;
    int h = (koff & 7) | ((koff & 16) >> 1);
    int lane = l15 | (g << 4);
    return ((rowtile * ktiles + kt) * 32 + lane) * 16 + h;
}

// A fragment gathered from f32 LDS (converted), leading dim ldb.
__device__ __forceinline__ v16h fragA_f32(const float* base, int ldb, int row0, int k0, int lane) {
    v16h f;
    const float* p = base + (row0 + (lane & 15)) * ldb + k0;
#pragma unroll
    for (int h = 0; h < 16; ++h) f[h] = (half_t)p[fkoff(lane, h)];
    return f;
}

// ---- prep kernel: swizzle f32 weights (batch x K x N) into fragment-ordered f16,
// zero-padded to Kp rows. dst layout: [b][kt][ct][lane][h].
__global__ void swizzle_w(const float* __restrict__ src, half_t* __restrict__ dst,
                          int K, int N, int Kp, int batch) {
    const int KT = Kp >> 5, CT = N >> 4;
    const size_t total = (size_t)batch * KT * CT * 512;
    for (size_t s = (size_t)blockIdx.x * blockDim.x + threadIdx.x; s < total;
         s += (size_t)gridDim.x * blockDim.x) {
        int h = (int)(s & 15), lane = (int)((s >> 4) & 31);
        size_t tile = s >> 9;
        int ct = (int)(tile % CT); size_t t2 = tile / CT;
        int kt = (int)(t2 % KT);  int b  = (int)(t2 / KT);
        int k = kt * 32 + fkoff(lane, h);
        int n = ct * 16 + (lane & 15);
        float v = (k < K) ? src[(size_t)b * K * N + (size_t)k * N + n] : 0.f;
        dst[s] = (half_t)v;
    }
}

// fb_block means into small LDS buffers.
__device__ void build_means(const float* sV, int lds, int ds, const float* pV, int dp,
                            float* muB, float* mdB, float* puB, float* pdB, int tid) {
    for (int idx = tid; idx < WPB * ds; idx += THREADS) {
        int w = idx / ds, c = idx - w * ds;
        float su = 0.f, sd = 0.f;
#pragma unroll
        for (int e = 0; e < NU; ++e) su += sV[(w * NE + e) * lds + c];
#pragma unroll
        for (int e = NU; e < NE; ++e) sd += sV[(w * NE + e) * lds + c];
        muB[idx] = su * 0.125f; mdB[idx] = sd * 0.125f;
    }
    for (int idx = tid; idx < WPB * NE * dp; idx += THREADS) {
        int c = idx % dp, j = (idx / dp) & 15, w = idx / (dp * NE);
        float su = 0.f, sd = 0.f;
#pragma unroll
        for (int i = 0; i < NU; ++i) su += pV[((w * NE + i) * NE + j) * dp + c];
#pragma unroll
        for (int i = NU; i < NE; ++i) sd += pV[((w * NE + i) * NE + j) * dp + c];
        puB[idx] = su * 0.125f; pdB[idx] = sd * 0.125f;
    }
}

// Fill fragment-ordered f16 blk, one aligned 8-column octet (16B LDS store) per step.
__device__ void build_blk(half_t* blk, const float* sV, int lds, int ds,
                          const float* muB, const float* mdB,
                          const float* puB, const float* pdB,
                          int dp, int ktiles, int tid) {
    const int width = 3 * ds + 2 * dp;
    const int octs  = (ktiles * 32) >> 3;
    for (int idx = tid; idx < ROWS * octs; idx += THREADS) {
        int row = idx / octs, c0 = (idx - row * octs) << 3;
        int w = row >> 4, j = row & 15;
        v8h ov;
#pragma unroll
        for (int u = 0; u < 8; ++u) {
            int c = c0 + u;
            float v;
            if (c < ds)               v = sV[row * lds + c];
            else if (c < 2 * ds)      v = muB[w * ds + (c - ds)];
            else if (c < 3 * ds)      v = mdB[w * ds + (c - 2 * ds)];
            else if (c < 3 * ds + dp) v = puB[(w * NE + j) * dp + (c - 3 * ds)];
            else if (c < width)       v = pdB[(w * NE + j) * dp + (c - 3 * ds - dp)];
            else                      v = 0.f;
            ov[u] = (half_t)v;
        }
        *(v8h*)(blk + blk_index(row, c0, ktiles)) = ov;
    }
}

// s-stream matmul: each wave owns BOTH row-tiles x 2 col-tiles (B loaded once),
// software-pipelined: next K-step fragments prefetched before current WMMAs.
__device__ void mm_s(const half_t* blkS, const half_t* Wsw, const float* bias,
                     float* sV, int ktiles, bool residual, int tid) {
    const int wave = tid >> 5, lane = tid & 31;
    const int cb = wave * 2;                  // col tiles 2w, 2w+1 of 16
    const int lofs = lane * 16;
    v8f acc0 = {}, acc1 = {}, acc2 = {}, acc3 = {};
    const half_t* aP0 = blkS + lofs;                       // rowtile 0
    const half_t* aP1 = blkS + (size_t)ktiles * 512 + lofs; // rowtile 1
    const half_t* bP  = Wsw + ((size_t)cb * 32) * 16 + lofs;
    v16h a0 = *(const v16h*)aP0;
    v16h a1 = *(const v16h*)aP1;
    v16h b0 = ldg_v16h(bP);
    v16h b1 = ldg_v16h(bP + 512);
    for (int kt = 1; kt < ktiles; ++kt) {
        v16h na0 = *(const v16h*)(aP0 + kt * 512);
        v16h na1 = *(const v16h*)(aP1 + kt * 512);
        v16h nb0 = ldg_v16h(bP + (size_t)kt * 16 * 512);
        v16h nb1 = ldg_v16h(bP + (size_t)kt * 16 * 512 + 512);
        acc0 = __builtin_amdgcn_wmma_f32_16x16x32_f16(false, a0, false, b0, (short)0, acc0, false, false);
        acc1 = __builtin_amdgcn_wmma_f32_16x16x32_f16(false, a0, false, b1, (short)0, acc1, false, false);
        acc2 = __builtin_amdgcn_wmma_f32_16x16x32_f16(false, a1, false, b0, (short)0, acc2, false, false);
        acc3 = __builtin_amdgcn_wmma_f32_16x16x32_f16(false, a1, false, b1, (short)0, acc3, false, false);
        a0 = na0; a1 = na1; b0 = nb0; b1 = nb1;
    }
    acc0 = __builtin_amdgcn_wmma_f32_16x16x32_f16(false, a0, false, b0, (short)0, acc0, false, false);
    acc1 = __builtin_amdgcn_wmma_f32_16x16x32_f16(false, a0, false, b1, (short)0, acc1, false, false);
    acc2 = __builtin_amdgcn_wmma_f32_16x16x32_f16(false, a1, false, b0, (short)0, acc2, false, false);
    acc3 = __builtin_amdgcn_wmma_f32_16x16x32_f16(false, a1, false, b1, (short)0, acc3, false, false);

    const int n15 = lane & 15;
    const int rb  = (lane & 16) >> 1;
    const int nA = cb * 16 + n15, nB = nA + 16;
    const float bvA = bias[nA], bvB = bias[nB];
#pragma unroll
    for (int v = 0; v < 8; ++v) {
        int r0 = rb + v, r1 = 16 + rb + v;
        float v0 = fast_tanh(acc0[v] + bvA);
        float v1 = fast_tanh(acc1[v] + bvB);
        float v2 = fast_tanh(acc2[v] + bvA);
        float v3 = fast_tanh(acc3[v] + bvB);
        if (residual) {
            v0 += sV[r0 * NSV + nA]; v1 += sV[r0 * NSV + nB];
            v2 += sV[r1 * NSV + nA]; v3 += sV[r1 * NSV + nB];
        }
        sV[r0 * NSV + nA] = v0; sV[r0 * NSV + nB] = v1;
        sV[r1 * NSV + nA] = v2; sV[r1 * NSV + nB] = v3;
    }
}

// p-stream matmul: (512x32)@(32x32); B fragments hoisted.
__device__ void mm_p(const float* pc, const half_t* Wsw, const float* bias,
                     float* pn, int tid) {
    const int wave = tid >> 5, lane = tid & 31;
    const v16h b0 = ldg_v16h(Wsw + lane * 16);
    const v16h b1 = ldg_v16h(Wsw + (32 + lane) * 16);
    const int n15 = lane & 15;
    const float bv0 = bias[n15], bv1 = bias[16 + n15];
#pragma unroll
    for (int rt = 0; rt < 4; ++rt) {
        int row0 = (wave * 4 + rt) * 16;
        v16h a = fragA_f32(pc, NPV, row0, 0, lane);
        v8f a0 = {}, a1 = {};
        a0 = __builtin_amdgcn_wmma_f32_16x16x32_f16(false, a, false, b0, (short)0, a0, false, false);
        a1 = __builtin_amdgcn_wmma_f32_16x16x32_f16(false, a, false, b1, (short)0, a1, false, false);
        int rbase = row0 + ((lane & 16) >> 1);
#pragma unroll
        for (int v = 0; v < 8; ++v) {
            int r = rbase + v;
            pn[r * NPV + n15]      = fast_tanh(a0[v] + bv0) + pc[r * NPV + n15];
            pn[r * NPV + 16 + n15] = fast_tanh(a1[v] + bv1) + pc[r * NPV + 16 + n15];
        }
    }
}

// Orbitals: 16 rows x 256 @ 256x128 per spin, env-scaled, into per-det 8x8 M[k][i].
__device__ void mm_orb(const float* sV, const half_t* Wu, const half_t* Wd,
                       const float* bu, const float* bd,
                       const float* env, float* orb, int tid) {
    const int wave = tid >> 5, lane = tid & 31;
    const int i16 = lane & 15;
#pragma unroll
    for (int spin = 0; spin < 2; ++spin) {
        const half_t* Wsw = spin ? Wd : Wu;
        const float*  bias = spin ? bd : bu;
        const int aRow = (i16 >> 3) * NE + spin * 8 + (i16 & 7);
        v8f acc = {};
        for (int ks = 0; ks < 8; ++ks) {
            v16h a;
            const float* p = sV + aRow * NSV + ks * 32;
#pragma unroll
            for (int h = 0; h < 16; ++h) a[h] = (half_t)p[fkoff(lane, h)];
            v16h b = ldg_v16h(Wsw + ((size_t)(ks * 8 + wave) * 32 + lane) * 16);
            acc = __builtin_amdgcn_wmma_f32_16x16x32_f16(false, a, false, b,
                                                         (short)0, acc, false, false);
        }
        int n = wave * 16 + i16;
        int k = n >> 4, d = n & 15;
        float bv = bias[n];
        int rbase = (lane & 16) >> 1;
#pragma unroll
        for (int v = 0; v < 8; ++v) {
            int rr = rbase + v;
            int w = rr >> 3, i = rr & 7;
            float e = env[w * NE + spin * 8 + i];
            orb[((spin * WPB + w) * NDET + d) * 64 + k * 8 + i] = (acc[v] + bv) * e;
        }
    }
}

__global__ void __launch_bounds__(THREADS, 1)
ansatz_kernel(const float* __restrict__ r,   const float* __restrict__ a,
              const float* __restrict__ pW0, const float* __restrict__ pb0,
              const float* __restrict__ sb0, const float* __restrict__ sb,
              const float* __restrict__ pb,  const float* __restrict__ vb,
              const float* __restrict__ wub, const float* __restrict__ wdb,
              const float* __restrict__ wfW, const half_t* __restrict__ ws,
              float* __restrict__ out) {
    extern __shared__ char smem[];
    half_t* blk   = (half_t*)smem;
    float*  sV    = (float*)(smem + size_t(BLK_H) * 2);
    float*  pA    = sV   + SV_F;
    float*  pB    = pA   + P_F;
    float*  pRaw  = pB   + P_F;
    float*  orb   = pRaw + PRAW_F;
    float*  muB   = orb  + ORB_F;
    float*  mdB   = muB  + WPB * NSV;
    float*  puB   = mdB  + WPB * NSV;
    float*  pdB   = puB  + WPB * NE * NPV;
    float*  env   = pdB  + WPB * NE * NPV;
    float*  rL    = env  + ROWS;
    float*  aL    = rL   + ROWS * 3;
    float*  dsign = aL   + NA * 3;
    float*  dld   = dsign + 64;

    const int tid = threadIdx.x;
    const int gw0 = blockIdx.x * WPB;

    // ---- Phase 1: coords, s_v0 embedding, env, pRaw ----
    for (int idx = tid; idx < ROWS * 3; idx += THREADS) rL[idx] = r[gw0 * NE * 3 + idx];
    for (int idx = tid; idx < NA * 3; idx += THREADS)   aL[idx] = a[idx];
    __syncthreads();

    if (tid < ROWS) {
        float rx = rL[tid * 3], ry = rL[tid * 3 + 1], rz = rL[tid * 3 + 2];
        float ev = 0.f;
#pragma unroll
        for (int at = 0; at < NA; ++at) {
            float dx = rx - aL[at * 3], dy = ry - aL[at * 3 + 1], dz = rz - aL[at * 3 + 2];
            float len = sqrtf(dx * dx + dy * dy + dz * dz);
            sV[tid * NSV + at * 4 + 0] = dx;
            sV[tid * NSV + at * 4 + 1] = dy;
            sV[tid * NSV + at * 4 + 2] = dz;
            sV[tid * NSV + at * 4 + 3] = len;
            ev += __expf(-len);
        }
        env[tid] = ev;
    }
    for (int idx = tid; idx < PROWS; idx += THREADS) {
        int w = idx >> 8, i = (idx >> 4) & 15, j = idx & 15;
        const float* ri = rL + (w * NE + i) * 3;
        const float* rj = rL + (w * NE + j) * 3;
        float dx = rj[0] - ri[0], dy = rj[1] - ri[1], dz = rj[2] - ri[2];
        float e = (i == j) ? 1.f : 0.f;
        float lx = dx + e, ly = dy + e, lz = dz + e;
        pRaw[idx * 4 + 0] = dx; pRaw[idx * 4 + 1] = dy; pRaw[idx * 4 + 2] = dz;
        pRaw[idx * 4 + 3] = sqrtf(lx * lx + ly * ly + lz * lz);
    }
    __syncthreads();

    // ---- Phase 2: layer 0 ----
    build_means(sV, NSV, 4 * NA, pRaw, 4, muB, mdB, puB, pdB, tid);
    __syncthreads();
    build_blk(blk, sV, NSV, 4 * NA, muB, mdB, puB, pdB, 4, 2, tid);
    __syncthreads();
    mm_s(blk, ws + WS_SW0, sb0, sV, 2, false, tid);
    for (int idx = tid; idx < PROWS * NPV; idx += THREADS) {   // p0: K=4 scalar
        int row = idx >> 5, n = idx & 31;
        float s = pb0[n];
#pragma unroll
        for (int k2 = 0; k2 < 4; ++k2) s += pRaw[row * 4 + k2] * pW0[k2 * NPV + n];
        pA[idx] = fast_tanh(s);
    }
    __syncthreads();

    // ---- Phase 3: residual layers ----
    float* pc = pA; float* pn = pB;
    for (int l = 0; l < NFB; ++l) {
        build_means(sV, NSV, NSV, pc, NPV, muB, mdB, puB, pdB, tid);
        __syncthreads();
        build_blk(blk, sV, NSV, NSV, muB, mdB, puB, pdB, NPV, KT_SV, tid);
        __syncthreads();
        mm_s(blk, ws + WS_SW + l * SW_LAYER, sb + l * NSV, sV, KT_SV, true, tid);
        mm_p(pc, ws + WS_PW + l * 1024, pb + l * NPV, pn, tid);
        __syncthreads();
        float* t = pc; pc = pn; pn = t;
    }

    // ---- Phase 4: final projection ----
    build_means(sV, NSV, NSV, pc, NPV, muB, mdB, puB, pdB, tid);
    __syncthreads();
    build_blk(blk, sV, NSV, NSV, muB, mdB, puB, pdB, NPV, KT_SV, tid);
    __syncthreads();
    mm_s(blk, ws + WS_VW, vb, sV, KT_SV, false, tid);
    __syncthreads();

    // ---- Phase 5: orbitals ----
    mm_orb(sV, ws + WS_WU, ws + WS_WD, wub, wdb, env, orb, tid);
    __syncthreads();

    // ---- Phase 6: 8x8 slogdet via pivoted LU, one matrix per thread ----
    if (tid < 2 * WPB * NDET) {
        const float* M = orb + tid * 64;
        float A[64];
#pragma unroll
        for (int x = 0; x < 64; ++x) A[x] = M[x];
        float sg = 1.f, ld = 0.f;
        for (int c = 0; c < 8; ++c) {
            int p = c; float mx = fabsf(A[c * 8 + c]);
            for (int r2 = c + 1; r2 < 8; ++r2) {
                float av = fabsf(A[r2 * 8 + c]);
                if (av > mx) { mx = av; p = r2; }
            }
            if (p != c) {
                for (int cc = 0; cc < 8; ++cc) {
                    float tv = A[c * 8 + cc]; A[c * 8 + cc] = A[p * 8 + cc]; A[p * 8 + cc] = tv;
                }
                sg = -sg;
            }
            float piv = A[c * 8 + c];
            if (piv < 0.f) sg = -sg;
            ld += logf(fabsf(piv));
            float inv = 1.f / piv;
            for (int r2 = c + 1; r2 < 8; ++r2) {
                float f = A[r2 * 8 + c] * inv;
                for (int cc = c + 1; cc < 8; ++cc) A[r2 * 8 + cc] -= f * A[c * 8 + cc];
            }
        }
        dsign[tid] = sg; dld[tid] = ld;
    }
    __syncthreads();

    // ---- Phase 7: logsumexp combine per walker ----
    if (tid < WPB) {
        float ldt[NDET], sgn[NDET], m = -1e30f;
#pragma unroll
        for (int d = 0; d < NDET; ++d) {
            ldt[d] = dld[tid * 16 + d] + dld[32 + tid * 16 + d];
            sgn[d] = dsign[tid * 16 + d] * dsign[32 + tid * 16 + d];
            m = fmaxf(m, ldt[d]);
        }
        float psi = 0.f;
#pragma unroll
        for (int d = 0; d < NDET; ++d) psi += sgn[d] * expf(ldt[d] - m) * wfW[d];
        out[gw0 + tid] = logf(fabsf(psi)) + m;
    }
}

extern "C" void kernel_launch(void* const* d_in, const int* in_sizes, int n_in,
                              void* d_out, int out_size, void* d_ws, size_t ws_size,
                              hipStream_t stream) {
    (void)in_sizes; (void)n_in; (void)out_size; (void)ws_size;
    const float* r   = (const float*)d_in[0];
    const float* a   = (const float*)d_in[1];
    const float* sW0 = (const float*)d_in[2];
    const float* sb0 = (const float*)d_in[3];
    const float* sW  = (const float*)d_in[4];
    const float* sb  = (const float*)d_in[5];
    const float* pW0 = (const float*)d_in[6];
    const float* pb0 = (const float*)d_in[7];
    const float* pW  = (const float*)d_in[8];
    const float* pb  = (const float*)d_in[9];
    const float* vW  = (const float*)d_in[10];
    const float* vb  = (const float*)d_in[11];
    const float* wuW = (const float*)d_in[12];
    const float* wub = (const float*)d_in[13];
    const float* wdW = (const float*)d_in[14];
    const float* wdb = (const float*)d_in[15];
    const float* wfW = (const float*)d_in[16];
    float* out = (float*)d_out;
    half_t* ws = (half_t*)d_ws;

    swizzle_w<<<64, THREADS, 0, stream>>>(sW0, ws + WS_SW0, 56, 256, 64, 1);
    swizzle_w<<<512, THREADS, 0, stream>>>(sW, ws + WS_SW, SVOUT, 256, SVOUT, 4);
    swizzle_w<<<256, THREADS, 0, stream>>>(vW, ws + WS_VW, SVOUT, 256, SVOUT, 1);
    swizzle_w<<<8, THREADS, 0, stream>>>(pW, ws + WS_PW, 32, 32, 32, 4);
    swizzle_w<<<64, THREADS, 0, stream>>>(wuW, ws + WS_WU, 256, 128, 256, 1);
    swizzle_w<<<64, THREADS, 0, stream>>>(wdW, ws + WS_WD, 256, 128, 256, 1);

    (void)hipFuncSetAttribute((const void*)ansatz_kernel,
                              hipFuncAttributeMaxDynamicSharedMemorySize,
                              (int)SMEM_BYTES);
    ansatz_kernel<<<2048 / WPB, THREADS, SMEM_BYTES, stream>>>(
        r, a, pW0, pb0, sb0, sb, pb, vb, wub, wdb, wfW, ws, out);
}
